// TransformerBlock_64476049047605
// MI455X (gfx1250) — matrix operand order repaired and checked by reading, not compiled, not run on hardware
//
#include <hip/hip_runtime.h>
#include <hip/hip_bf16.h>

// ---------------------------------------------------------------- constants
constexpr int B_   = 2;
constexpr int S_   = 2048;
constexpr int D_   = 1024;
constexpr int NH_  = 16;
constexpr int FF_  = 4096;
constexpr int HD_  = 64;           // D/NH
constexpr int MTOT = B_ * S_;      // 4096 rows
constexpr float SCALE_ = 0.125f;   // HD^-0.5
constexpr float EPS_   = 1e-5f;

typedef __attribute__((ext_vector_type(8)))  float  v8f;
typedef __attribute__((ext_vector_type(4)))  __bf16 v4bf;
typedef __attribute__((ext_vector_type(8)))  __bf16 v8bf;
typedef __attribute__((ext_vector_type(16))) __bf16 v16bf;

// bf16 WMMA: D(16x16,f32) = A(16x32,bf16) * B(32x16,bf16) + C
__device__ __forceinline__ v8f wmma_bf16(v16bf a, v16bf b, v8f c) {
  return __builtin_amdgcn_wmma_f32_16x16x32_bf16(
      false, a, false, b, (short)0, c, false, false);
}

// A-fragment: row-major row m; lane half hi: k in [hi*16, hi*16+16)
__device__ __forceinline__ v16bf load_afrag(const __bf16* p) {
  const v8bf c0 = *(const v8bf*)(p);
  const v8bf c1 = *(const v8bf*)(p + 8);
  return __builtin_shufflevector(c0, c1, 0,1,2,3,4,5,6,7,8,9,10,11,12,13,14,15);
}
// B-fragment from k-major row: k in [hi*16, hi*16+16) -> one contiguous 32B load

// ---------------------------------------------------------------- layernorm (fp32 in -> bf16 out)
__global__ void layernorm_kernel(const float* __restrict__ x,
                                 const float* __restrict__ g,
                                 const float* __restrict__ bta,
                                 __bf16* __restrict__ out) {
  const int t    = threadIdx.x;
  const int lane = t & 31;
  const int wave = t >> 5;
  const size_t row = blockIdx.x;
  const float4* xr = (const float4*)(x + row * (size_t)D_);
  float4 v = xr[t];
  float s  = v.x + v.y + v.z + v.w;
  float s2 = v.x * v.x + v.y * v.y + v.z * v.z + v.w * v.w;
  #pragma unroll
  for (int off = 16; off > 0; off >>= 1) {
    s  += __shfl_xor(s,  off, 32);
    s2 += __shfl_xor(s2, off, 32);
  }
  __shared__ float ps[8], ps2[8];
  if (lane == 0) { ps[wave] = s; ps2[wave] = s2; }
  __syncthreads();
  if (t == 0) {
    float a = 0.f, b2 = 0.f;
    #pragma unroll
    for (int i = 0; i < 8; ++i) { a += ps[i]; b2 += ps2[i]; }
    ps[0] = a; ps2[0] = b2;
  }
  __syncthreads();
  const float mean = ps[0] * (1.0f / D_);
  const float var  = ps2[0] * (1.0f / D_) - mean * mean;
  const float inv  = rsqrtf(var + EPS_);
  const float4 gg = ((const float4*)g)[t];
  const float4 bb = ((const float4*)bta)[t];
  v4bf o;
  o[0] = (__bf16)((v.x - mean) * inv * gg.x + bb.x);
  o[1] = (__bf16)((v.y - mean) * inv * gg.y + bb.y);
  o[2] = (__bf16)((v.z - mean) * inv * gg.z + bb.z);
  o[3] = (__bf16)((v.w - mean) * inv * gg.w + bb.w);
  *(v4bf*)(out + row * (size_t)D_ + t * 4) = o;
}

// ---------------------------------------------------------------- weight transpose fp32[R,C] -> bf16[C,R]
__global__ void transpose_wt_kernel(const float* __restrict__ src,
                                    __bf16* __restrict__ dst, int R, int C) {
  __shared__ float tile[32][33];
  const int c0 = blockIdx.x * 32, r0 = blockIdx.y * 32;
  const int xx = threadIdx.x, yy = threadIdx.y;   // (32, 8)
  #pragma unroll
  for (int i = 0; i < 32; i += 8)
    tile[yy + i][xx] = src[(size_t)(r0 + yy + i) * C + c0 + xx];
  __syncthreads();
  #pragma unroll
  for (int i = 0; i < 32; i += 8)
    dst[(size_t)(c0 + yy + i) * R + r0 + xx] = (__bf16)tile[xx][yy + i];
}

// ---------------------------------------------------------------- V transpose: kv_bf V-half -> vT[b,h,d,m]
// Uses CDNA5 async global->LDS copies (ASYNCcnt) for the staging step.
__global__ void vtrans_kernel(const __bf16* __restrict__ kvb, __bf16* __restrict__ vT) {
  __shared__ __bf16 tile[64 * 72];
  const int bh = blockIdx.y, b = bh >> 4, h = bh & 15;
  const int m0 = blockIdx.x * 64;
  const int t  = threadIdx.x;
  const int r  = t >> 2;            // 0..63
  const int c0 = (t & 3) * 16;      // 0..48
  const __bf16* src = kvb + (size_t)(b * S_ + m0 + r) * (2 * D_) + D_ + h * HD_ + c0;
  {
    const unsigned int ldsoff = (unsigned int)(uintptr_t)(&tile[r * 72 + c0]);
    const unsigned long long ga = (unsigned long long)(uintptr_t)src;
    asm volatile("global_load_async_to_lds_b128 %0, %1, off\n\t"
                 "global_load_async_to_lds_b128 %2, %3, off"
                 :: "v"(ldsoff), "v"(ga), "v"(ldsoff + 16u), "v"(ga + 16ull)
                 : "memory");
    asm volatile("s_wait_asynccnt 0x0" ::: "memory");
  }
  __syncthreads();
  __bf16* dst = vT + ((size_t)bh * HD_ + r) * S_ + m0 + c0;   // r as d, c0 as m-seg
  v8bf o0, o1;
  #pragma unroll
  for (int i = 0; i < 8; ++i) {
    o0[i] = tile[(c0 + i) * 72 + r];
    o1[i] = tile[(c0 + 8 + i) * 72 + r];
  }
  *(v8bf*)dst       = o0;
  *(v8bf*)(dst + 8) = o1;
}

// ---------------------------------------------------------------- dense GEMM (bf16 x bf16^T -> fp32/bf16)
// A[M,K] bf16 row-major, BT[N,K] bf16 row-major. K % 64 == 0.
// 8 waves; wave tile 32(m) x 64(n); block tile 128 x 128.
// Ping-pong unroll-by-2 over k: no register copies, loads covered by a WMMA block.
__global__ __launch_bounds__(256, 1)
void gemm_bf16_kernel(const __bf16* __restrict__ A,
                      const __bf16* __restrict__ BT,
                      const float* __restrict__ bias,
                      const float* __restrict__ res,
                      float* __restrict__ Cf,
                      __bf16* __restrict__ Cb,
                      int K, int N, int relu) {
  const int lane = threadIdx.x & 31;
  const int wave = threadIdx.x >> 5;
  const int l16  = lane & 15;
  const int hi   = lane >> 4;
  const int row0 = blockIdx.y * 128 + (wave & 3) * 32;
  const int col0 = blockIdx.x * 128 + (wave >> 2) * 64;

  const __bf16* Ap0 = A + (size_t)(row0 + l16) * K + hi * 16;
  const __bf16* Ap1 = Ap0 + (size_t)16 * K;
  const __bf16* Bp[4];
  #pragma unroll
  for (int j = 0; j < 4; ++j)
    Bp[j] = BT + (size_t)(col0 + 16 * j + l16) * K + hi * 16;

  v8f acc[2][4] = {};
  v16bf a0A, a1A, bA[4], a0B, a1B, bB[4];
  a0A = load_afrag(Ap0);
  a1A = load_afrag(Ap1);
  #pragma unroll
  for (int j = 0; j < 4; ++j) bA[j] = *(const v16bf*)(Bp[j]);

  for (int k0 = 0; k0 < K; k0 += 64) {
    const int kb = k0 + 32;
    a0B = load_afrag(Ap0 + kb);
    a1B = load_afrag(Ap1 + kb);
    #pragma unroll
    for (int j = 0; j < 4; ++j) bB[j] = *(const v16bf*)(Bp[j] + kb);
    #pragma unroll
    for (int j = 0; j < 4; ++j) {
      acc[0][j] = wmma_bf16(a0A, bA[j], acc[0][j]);
      acc[1][j] = wmma_bf16(a1A, bA[j], acc[1][j]);
    }
    const int ka = k0 + 64;
    if (ka < K) {
      a0A = load_afrag(Ap0 + ka);
      a1A = load_afrag(Ap1 + ka);
      #pragma unroll
      for (int j = 0; j < 4; ++j) bA[j] = *(const v16bf*)(Bp[j] + ka);
    }
    #pragma unroll
    for (int j = 0; j < 4; ++j) {
      acc[0][j] = wmma_bf16(a0B, bB[j], acc[0][j]);
      acc[1][j] = wmma_bf16(a1B, bB[j], acc[1][j]);
    }
  }

  const int rhi = hi * 8;
  #pragma unroll
  for (int i = 0; i < 2; ++i)
    #pragma unroll
    for (int j = 0; j < 4; ++j) {
      const int col = col0 + 16 * j + l16;
      const float badd = bias ? bias[col] : 0.0f;
      #pragma unroll
      for (int r = 0; r < 8; ++r) {
        const size_t idx = (size_t)(row0 + 16 * i + rhi + r) * N + col;
        float v = acc[i][j][r] + badd;
        if (res)  v += res[idx];
        if (relu) v = fmaxf(v, 0.0f);
        if (Cf) Cf[idx] = v;
        if (Cb) Cb[idx] = (__bf16)v;
      }
    }
}

// ---------------------------------------------------------------- scores: attn[b,n,m,h] = SCALE * q.k
// grid: x = S/64 (m), y = S/128 (n), z = B*NH; wave tile 32x32, K=64 (2 steps)
__global__ void attn_scores_kernel(const __bf16* __restrict__ q,
                                   const __bf16* __restrict__ kvb,
                                   float* __restrict__ attn) {
  const int lane = threadIdx.x & 31;
  const int wave = threadIdx.x >> 5;
  const int l16  = lane & 15;
  const int hi   = lane >> 4;
  const int bh = blockIdx.z, b = bh >> 4, h = bh & 15;
  const int n0 = blockIdx.y * 128 + (wave & 3) * 32;
  const int m0 = blockIdx.x * 64  + (wave >> 2) * 32;

  const __bf16* qb = q   + (size_t)b * S_ * D_ + h * HD_;
  const __bf16* kb = kvb + (size_t)b * S_ * (2 * D_) + h * HD_;    // K half
  const __bf16* Ap0 = qb + (size_t)(n0 + l16) * D_ + hi * 16;
  const __bf16* Ap1 = Ap0 + (size_t)16 * D_;
  const __bf16* Bp0 = kb + (size_t)(m0 + l16) * (2 * D_) + hi * 16;
  const __bf16* Bp1 = Bp0 + (size_t)16 * (2 * D_);

  v8f acc[2][2] = {};
  #pragma unroll
  for (int k0 = 0; k0 < HD_; k0 += 32) {
    const v16bf a0 = load_afrag(Ap0 + k0);
    const v16bf a1 = load_afrag(Ap1 + k0);
    const v16bf b0 = *(const v16bf*)(Bp0 + k0);
    const v16bf b1 = *(const v16bf*)(Bp1 + k0);
    acc[0][0] = wmma_bf16(a0, b0, acc[0][0]);
    acc[0][1] = wmma_bf16(a0, b1, acc[0][1]);
    acc[1][0] = wmma_bf16(a1, b0, acc[1][0]);
    acc[1][1] = wmma_bf16(a1, b1, acc[1][1]);
  }
  const int rhi = hi * 8;
  #pragma unroll
  for (int i = 0; i < 2; ++i)
    #pragma unroll
    for (int j = 0; j < 2; ++j) {
      const int m = m0 + 16 * j + l16;
      #pragma unroll
      for (int r = 0; r < 8; ++r) {
        const int n = n0 + 16 * i + rhi + r;
        attn[(((size_t)(b * S_ + n)) * S_ + m) * NH_ + h] = acc[i][j][r] * SCALE_;
      }
    }
}

// ---------------------------------------------------------------- softmax (fp32 in place) + bf16 transposed mirror
__global__ void softmax_kernel(float* __restrict__ attn, __bf16* __restrict__ attnT) {
  const int t  = threadIdx.x;
  const int h  = t & 15;
  const int lm = t >> 4;
  const int row = blockIdx.x;            // b*S + n
  const int b = row >> 11, n = row & (S_ - 1);
  float* base = attn + (size_t)row * S_ * NH_;

  float f[64];
  float lmax = -3.4e38f;
  #pragma unroll
  for (int j = 0; j < 64; ++j) {
    f[j] = base[(size_t)(lm + 32 * j) * NH_ + h];
    lmax = fmaxf(lmax, f[j]);
  }
  __shared__ float red[32 * 16];
  __shared__ float fin[16];
  red[lm * 16 + h] = lmax;
  __syncthreads();
  if (t < 16) {
    float m = red[h];
    for (int i = 1; i < 32; ++i) m = fmaxf(m, red[i * 16 + h]);
    fin[h] = m;
  }
  __syncthreads();
  const float gm = fin[h];
  float ls = 0.f;
  #pragma unroll
  for (int j = 0; j < 64; ++j) { f[j] = __expf(f[j] - gm); ls += f[j]; }
  __syncthreads();
  red[lm * 16 + h] = ls;
  __syncthreads();
  if (t < 16) {
    float sum = 0.f;
    for (int i = 0; i < 32; ++i) sum += red[i * 16 + h];
    fin[h] = 1.0f / sum;
  }
  __syncthreads();
  const float inv = fin[h];
  #pragma unroll
  for (int j = 0; j < 64; ++j) {
    f[j] *= inv;
    base[(size_t)(lm + 32 * j) * NH_ + h] = f[j];
  }

  // bf16 mirror in (b,h,n,m): LDS transpose, 8 head-planes per pass
  __shared__ __bf16 sbt[8 * 2056];
  const int w = t >> 5, lane = t & 31;
  #pragma unroll
  for (int p = 0; p < 2; ++p) {
    __syncthreads();
    if ((h >> 3) == p) {
      #pragma unroll
      for (int j = 0; j < 64; ++j)
        sbt[(h & 7) * 2056 + lm + 32 * j] = (__bf16)f[j];
    }
    __syncthreads();
    const int hp = p * 8 + (w >> 1);
    const uint2* sp = (const uint2*)(sbt + (w >> 1) * 2056);
    uint2* gp = (uint2*)(attnT + ((size_t)(b * NH_ + hp) * S_ + n) * S_);
    const int seg = (w & 1) * 256 + lane;
    #pragma unroll
    for (int i = 0; i < 8; ++i)
      gp[seg + 32 * i] = sp[seg + 32 * i];
  }
}

// ---------------------------------------------------------------- attn @ v (k-contiguous bf16 operands)
// grid: y = S/128 (n), z = B*NH; wave tile 32(n) x 32(d); ping-pong unroll-by-2
__global__ __launch_bounds__(256, 1)
void attn_v_kernel(const __bf16* __restrict__ attnT,
                   const __bf16* __restrict__ vT,
                   __bf16* __restrict__ ov) {
  const int lane = threadIdx.x & 31;
  const int wave = threadIdx.x >> 5;
  const int l16  = lane & 15;
  const int hi   = lane >> 4;
  const int bh = blockIdx.z, b = bh >> 4, h = bh & 15;
  const int n0 = blockIdx.y * 128 + (wave & 3) * 32;
  const int d0 = (wave >> 2) * 32;

  const __bf16* ab = attnT + (size_t)bh * S_ * S_;
  const __bf16* vb = vT + (size_t)bh * HD_ * S_;
  const __bf16* Ap0 = ab + (size_t)(n0 + l16) * S_ + hi * 16;
  const __bf16* Ap1 = Ap0 + (size_t)16 * S_;
  const __bf16* Bp0 = vb + (size_t)(d0 + l16) * S_ + hi * 16;
  const __bf16* Bp1 = Bp0 + (size_t)16 * S_;

  v8f acc[2][2] = {};
  v16bf a0A, a1A, b0A, b1A, a0B, a1B, b0B, b1B;
  a0A = load_afrag(Ap0);
  a1A = load_afrag(Ap1);
  b0A = *(const v16bf*)(Bp0);
  b1A = *(const v16bf*)(Bp1);
  for (int k0 = 0; k0 < S_; k0 += 64) {
    const int kb = k0 + 32;
    a0B = load_afrag(Ap0 + kb);
    a1B = load_afrag(Ap1 + kb);
    b0B = *(const v16bf*)(Bp0 + kb);
    b1B = *(const v16bf*)(Bp1 + kb);
    acc[0][0] = wmma_bf16(a0A, b0A, acc[0][0]);
    acc[0][1] = wmma_bf16(a0A, b1A, acc[0][1]);
    acc[1][0] = wmma_bf16(a1A, b0A, acc[1][0]);
    acc[1][1] = wmma_bf16(a1A, b1A, acc[1][1]);
    const int ka = k0 + 64;
    if (ka < S_) {
      a0A = load_afrag(Ap0 + ka);
      a1A = load_afrag(Ap1 + ka);
      b0A = *(const v16bf*)(Bp0 + ka);
      b1A = *(const v16bf*)(Bp1 + ka);
    }
    acc[0][0] = wmma_bf16(a0B, b0B, acc[0][0]);
    acc[0][1] = wmma_bf16(a0B, b1B, acc[0][1]);
    acc[1][0] = wmma_bf16(a1B, b0B, acc[1][0]);
    acc[1][1] = wmma_bf16(a1B, b1B, acc[1][1]);
  }
  const int rhi = hi * 8;
  __bf16* ob = ov + (size_t)b * S_ * D_ + h * HD_;
  #pragma unroll
  for (int i = 0; i < 2; ++i)
    #pragma unroll
    for (int j = 0; j < 2; ++j) {
      const int d = d0 + 16 * j + l16;
      #pragma unroll
      for (int r = 0; r < 8; ++r) {
        const int n = n0 + 16 * i + rhi + r;
        ob[(size_t)n * D_ + d] = (__bf16)acc[i][j][r];
      }
    }
}

// ---------------------------------------------------------------- launcher
extern "C" void kernel_launch(void* const* d_in, const int* in_sizes, int n_in,
                              void* d_out, int out_size, void* d_ws, size_t ws_size,
                              hipStream_t stream) {
  const float* x     = (const float*)d_in[0];
  const float* ln1_g = (const float*)d_in[1];
  const float* ln1_b = (const float*)d_in[2];
  const float* wq    = (const float*)d_in[3];
  const float* wkv   = (const float*)d_in[4];
  const float* wo    = (const float*)d_in[5];
  const float* bo    = (const float*)d_in[6];
  const float* ln2_g = (const float*)d_in[7];
  const float* ln2_b = (const float*)d_in[8];
  const float* w1    = (const float*)d_in[9];
  const float* b1    = (const float*)d_in[10];
  const float* w2    = (const float*)d_in[11];
  const float* b2    = (const float*)d_in[12];

  float* out0 = (float*)d_out;                  // (B,S,D) fp32
  float* attn = out0 + (size_t)MTOT * D_;       // (B,S,S,NH) fp32

  char* wp = (char*)d_ws;
  auto take = [&](size_t bytes) -> void* {
    void* r = (void*)wp;
    wp += (bytes + 255) & ~(size_t)255;
    return r;
  };
  __bf16* attnT = (__bf16*)take((size_t)B_ * NH_ * S_ * S_ * 2);   // 256MB
  __bf16* h_bf  = (__bf16*)take((size_t)MTOT * D_ * 2);
  __bf16* q_bf  = (__bf16*)take((size_t)MTOT * D_ * 2);
  __bf16* kv_bf = (__bf16*)take((size_t)MTOT * 2 * D_ * 2);
  __bf16* vTp   = (__bf16*)take((size_t)B_ * NH_ * HD_ * S_ * 2);
  __bf16* ov_bf = (__bf16*)take((size_t)MTOT * D_ * 2);
  __bf16* h2_bf = (__bf16*)take((size_t)MTOT * D_ * 2);
  __bf16* ff_bf = (__bf16*)take((size_t)MTOT * FF_ * 2);
  float*  x1    = (float*) take((size_t)MTOT * D_ * 4);
  __bf16* wqT   = (__bf16*)take((size_t)D_ * D_ * 2);
  __bf16* wkvT  = (__bf16*)take((size_t)D_ * 2 * D_ * 2);
  __bf16* woT   = (__bf16*)take((size_t)D_ * D_ * 2);
  __bf16* w1T   = (__bf16*)take((size_t)D_ * FF_ * 2);
  __bf16* w2T   = (__bf16*)take((size_t)FF_ * D_ * 2);

  const dim3 blk(256);
  const dim3 tblk(32, 8);

  // weight transposes (fp32 -> bf16 W^T), L2-resident for the whole launch
  transpose_wt_kernel<<<dim3(D_ / 32, D_ / 32), tblk, 0, stream>>>(wq, wqT, D_, D_);
  transpose_wt_kernel<<<dim3(2 * D_ / 32, D_ / 32), tblk, 0, stream>>>(wkv, wkvT, D_, 2 * D_);
  transpose_wt_kernel<<<dim3(D_ / 32, D_ / 32), tblk, 0, stream>>>(wo, woT, D_, D_);
  transpose_wt_kernel<<<dim3(FF_ / 32, D_ / 32), tblk, 0, stream>>>(w1, w1T, D_, FF_);
  transpose_wt_kernel<<<dim3(D_ / 32, FF_ / 32), tblk, 0, stream>>>(w2, w2T, FF_, D_);

  // 1. h = layer_norm(x)            (bf16)
  layernorm_kernel<<<MTOT, blk, 0, stream>>>(x, ln1_g, ln1_b, h_bf);
  // 2. q = h @ wq                   (bf16 only)
  gemm_bf16_kernel<<<dim3(D_ / 128, MTOT / 128), blk, 0, stream>>>(
      h_bf, wqT, nullptr, nullptr, nullptr, q_bf, D_, D_, 0);
  // 3. kv = h @ wkv                 (bf16 only)
  gemm_bf16_kernel<<<dim3(2 * D_ / 128, MTOT / 128), blk, 0, stream>>>(
      h_bf, wkvT, nullptr, nullptr, nullptr, kv_bf, D_, 2 * D_, 0);
  // 4. vT[b,h,d,m] from kv V-half   (async global->LDS staging)
  vtrans_kernel<<<dim3(S_ / 64, B_ * NH_), blk, 0, stream>>>(kv_bf, vTp);
  // 5. scores -> attn fp32 (scaled)
  attn_scores_kernel<<<dim3(S_ / 64, S_ / 128, B_ * NH_), blk, 0, stream>>>(q_bf, kv_bf, attn);
  // 6. softmax in place + bf16 transposed mirror
  softmax_kernel<<<MTOT, dim3(512), 0, stream>>>(attn, attnT);
  // 7. ov = attn @ v                (bf16 only)
  attn_v_kernel<<<dim3(1, S_ / 128, B_ * NH_), blk, 0, stream>>>(attnT, vTp, ov_bf);
  // 8. x1 = ov @ wo + bo + x        (fp32)
  gemm_bf16_kernel<<<dim3(D_ / 128, MTOT / 128), blk, 0, stream>>>(
      ov_bf, woT, bo, x, x1, nullptr, D_, D_, 0);
  // 9. h2 = layer_norm(x1)          (bf16)
  layernorm_kernel<<<MTOT, blk, 0, stream>>>(x1, ln2_g, ln2_b, h2_bf);
  // 10. ff = relu(h2 @ w1 + b1)     (bf16 only)
  gemm_bf16_kernel<<<dim3(FF_ / 128, MTOT / 128), blk, 0, stream>>>(
      h2_bf, w1T, b1, nullptr, nullptr, ff_bf, D_, FF_, 1);
  // 11. out0 = ff @ w2 + b2 + x1    (fp32)
  gemm_bf16_kernel<<<dim3(D_ / 128, MTOT / 128), blk, 0, stream>>>(
      ff_bf, w2T, b2, x1, out0, nullptr, FF_, D_, 0);
}
